// HashedLinear_28132035789294
// MI455X (gfx1250) — matrix operand order repaired and checked by reading, not compiled, not run on hardware
//
#include <hip/hip_runtime.h>
#include <stdint.h>

// ---------------------------------------------------------------------------
// HashedLinear on MI455X (gfx1250):
//   out[t,o] = sum_k x[t,k] * weight[idx[o*K+k]] + bias[o]
// Strategy: gather-pack W into bf16 hi/lo planes (workspace, 64 MB), then a
// split-precision bf16 WMMA GEMM (hi*hi + hi*lo + lo*hi, fp32 accum) using
// v_wmma_f32_16x16x32_bf16.  Compute bound (~860 FLOP/B); L2 (192MB) holds
// the whole working set.  W tiles stream into LDS via the gfx1250 async
// global->LDS path (ASYNCcnt) with LDS double buffering; X is split to
// bf16 hi/lo in registers while staging into LDS.
// ---------------------------------------------------------------------------

#define TOKENS 8192
#define INF    4096   // in_features  (K)
#define OUTF   4096   // out_features (N)

#define BM 128        // block tile M
#define BN 128        // block tile N
#define BK 32         // K step (one WMMA K)
#define LDS_STRIDE 40 // 32 + 8 pad (uint16 units) -> 80B row stride, 16B aligned

#if __has_builtin(__builtin_amdgcn_global_load_async_to_lds_b128)
#define HAVE_ASYNC 1
#else
#define HAVE_ASYNC 0
#endif

#if HAVE_ASYNC
#if __has_builtin(__builtin_amdgcn_s_wait_asynccnt)
#define WAIT_ASYNC(n) __builtin_amdgcn_s_wait_asynccnt((unsigned short)(n))
#else
#define WAIT_ASYNC(n) asm volatile("s_wait_asynccnt %0" ::"i"(n) : "memory")
#endif
#endif

typedef __attribute__((ext_vector_type(16))) __bf16 v16bf;
typedef __attribute__((ext_vector_type(8)))  __bf16 v8bf;
typedef __attribute__((ext_vector_type(8)))  float  v8f;
typedef __attribute__((ext_vector_type(4)))  int    v4i;

typedef __attribute__((address_space(1))) v4i* gv4i_ptr;  // global int4*
typedef __attribute__((address_space(3))) v4i* lv4i_ptr;  // LDS    int4*

// ---- bf16 helpers ----------------------------------------------------------
__device__ __forceinline__ uint16_t f32_to_bf16_rne(float f) {
    uint32_t u = __float_as_uint(f);
    uint32_t r = u + 0x7FFFu + ((u >> 16) & 1u);
    return (uint16_t)(r >> 16);
}
__device__ __forceinline__ float bf16_bits_to_f32(uint16_t h) {
    return __uint_as_float(((uint32_t)h) << 16);
}
__device__ __forceinline__ void split_bf16(float f, uint16_t& h, uint16_t& l) {
    h = f32_to_bf16_rne(f);
    float fl = f - bf16_bits_to_f32(h);   // exact in fp32
    l = f32_to_bf16_rne(fl);
}

// ---- fragment loads from LDS (ISA 7.12.2 layouts, wave32) ------------------
// A 16x32 bf16: lane m holds K in {8h..8h+7} (vgpr0-3) and {16+8h..16+8h+7}.
__device__ __forceinline__ v16bf load_afrag(const uint16_t* s, int row, int half) {
    const v8bf a0 = *reinterpret_cast<const v8bf*>(s + row * LDS_STRIDE + 8 * half);
    const v8bf a1 = *reinterpret_cast<const v8bf*>(s + row * LDS_STRIDE + 16 + 8 * half);
    v16bf r;
#pragma unroll
    for (int i = 0; i < 8; ++i) { r[i] = a0[i]; r[i + 8] = a1[i]; }
    return r;
}
// B 32x16 bf16: lane n holds K in {8h..8h+7} and {16+8h..16+8h+7}.
__device__ __forceinline__ v16bf load_bfrag(const uint16_t* s, int row, int half) {
    const v8bf b0 = *reinterpret_cast<const v8bf*>(s + row * LDS_STRIDE + 8 * half);
    const v8bf b1 = *reinterpret_cast<const v8bf*>(s + row * LDS_STRIDE + 16 + 8 * half);
    v16bf r;
#pragma unroll
    for (int i = 0; i < 8; ++i) { r[i] = b0[i]; r[i + 8] = b1[i]; }
    return r;
}

// ---------------------------------------------------------------------------
// Kernel 1: gather bins -> dense bf16 hi/lo weight planes [OUTF, INF]
// ---------------------------------------------------------------------------
__global__ __launch_bounds__(256)
void pack_weights(const float* __restrict__ bins,
                  const int*   __restrict__ idx,
                  uint16_t* __restrict__ whi,
                  uint16_t* __restrict__ wlo) {
    const size_t i = ((size_t)blockIdx.x * 256 + threadIdx.x) * 4;
    const int4 id = *reinterpret_cast<const int4*>(idx + i);
    uint16_t h0, h1, h2, h3, l0, l1, l2, l3;
    split_bf16(bins[id.x], h0, l0);
    split_bf16(bins[id.y], h1, l1);
    split_bf16(bins[id.z], h2, l2);
    split_bf16(bins[id.w], h3, l3);
    uint2 ph, pl;
    ph.x = (uint32_t)h0 | ((uint32_t)h1 << 16);
    ph.y = (uint32_t)h2 | ((uint32_t)h3 << 16);
    pl.x = (uint32_t)l0 | ((uint32_t)l1 << 16);
    pl.y = (uint32_t)l2 | ((uint32_t)l3 << 16);
    *reinterpret_cast<uint2*>(whi + i) = ph;
    *reinterpret_cast<uint2*>(wlo + i) = pl;
}

// ---------------------------------------------------------------------------
// Kernel 2: GEMM  out = X * W^T + bias  via bf16x2 split WMMA
//   block = 256 threads (8 waves), tile 128(M) x 128(N), K-step 32
//   wave grid 4(M) x 2(N), wave tile 32x64 = 2x4 fragments of 16x16
//   B planes double-buffered in LDS, filled by async global->LDS loads.
// ---------------------------------------------------------------------------
__global__ __launch_bounds__(256)
void hashed_gemm(const float*    __restrict__ x,
                 const uint16_t* __restrict__ whi,
                 const uint16_t* __restrict__ wlo,
                 const float*    __restrict__ bias,
                 float*          __restrict__ out) {
    __shared__ uint16_t sAhi[BM * LDS_STRIDE];
    __shared__ uint16_t sAlo[BM * LDS_STRIDE];
    __shared__ uint16_t sBhi[2][BN * LDS_STRIDE];
    __shared__ uint16_t sBlo[2][BN * LDS_STRIDE];

    const int tid   = threadIdx.x;
    const int mBase = blockIdx.y * BM;   // token tile
    const int nBase = blockIdx.x * BN;   // out-feature tile
    const int wave  = tid >> 5;
    const int lane  = tid & 31;
    const int half  = lane >> 4;
    const int l16   = lane & 15;
    const int wm    = (wave & 3) * 32;   // wave M offset in tile
    const int wn    = (wave >> 2) * 64;  // wave N offset in tile

    // --- global-load index maps (per K-step) ---
    // X tile: 128 rows x 32 fp32 = 1024 float4 -> 4 per thread
    int xrow[4], xc4[4];
#pragma unroll
    for (int i = 0; i < 4; ++i) {
        const int lin = tid + i * 256;
        xrow[i] = lin >> 3;       // 8 float4 per row
        xc4[i]  = lin & 7;
    }
    // W tile: 128 rows x 32 bf16 = 512 16B-chunks -> 2 per thread (per plane)
    int wrow[2], wc8[2];
#pragma unroll
    for (int i = 0; i < 2; ++i) {
        const int lin = tid + i * 256;
        wrow[i] = lin >> 2;       // 4 chunks per row
        wc8[i]  = lin & 3;
    }

    float4 xr[4];

    auto load_x_regs = [&](int k0) {
#pragma unroll
        for (int i = 0; i < 4; ++i)
            xr[i] = *reinterpret_cast<const float4*>(
                x + (size_t)(mBase + xrow[i]) * INF + k0 + xc4[i] * 4);
    };

    auto store_a_lds = [&]() {
#pragma unroll
        for (int i = 0; i < 4; ++i) {
            uint16_t h[4], l[4];
            split_bf16(xr[i].x, h[0], l[0]);
            split_bf16(xr[i].y, h[1], l[1]);
            split_bf16(xr[i].z, h[2], l[2]);
            split_bf16(xr[i].w, h[3], l[3]);
            const int base = xrow[i] * LDS_STRIDE + xc4[i] * 4;
            uint2 ph, pl;
            ph.x = (uint32_t)h[0] | ((uint32_t)h[1] << 16);
            ph.y = (uint32_t)h[2] | ((uint32_t)h[3] << 16);
            pl.x = (uint32_t)l[0] | ((uint32_t)l[1] << 16);
            pl.y = (uint32_t)l[2] | ((uint32_t)l[3] << 16);
            *reinterpret_cast<uint2*>(sAhi + base) = ph;
            *reinterpret_cast<uint2*>(sAlo + base) = pl;
        }
    };

#if HAVE_ASYNC
    // Fire-and-forget global->LDS copies of a B k-slab (ASYNCcnt-tracked).
    // Builtin signature (from hipcc diagnostic): (v4i AS1*, v4i AS3*, Ii, Ii)
    auto issue_async_b = [&](int k0, int stage) {
#pragma unroll
        for (int i = 0; i < 2; ++i) {
            const size_t goff = (size_t)(nBase + wrow[i]) * INF + k0 + wc8[i] * 8;
            const int    lbase = wrow[i] * LDS_STRIDE + wc8[i] * 8;
            __builtin_amdgcn_global_load_async_to_lds_b128(
                (gv4i_ptr)(whi + goff),
                (lv4i_ptr)(&sBhi[stage][lbase]),
                0, 0);
            __builtin_amdgcn_global_load_async_to_lds_b128(
                (gv4i_ptr)(wlo + goff),
                (lv4i_ptr)(&sBlo[stage][lbase]),
                0, 0);
        }
    };
#else
    uint4 whr[2], wlr[2];
    auto load_b_regs = [&](int k0) {
#pragma unroll
        for (int i = 0; i < 2; ++i) {
            const size_t off = (size_t)(nBase + wrow[i]) * INF + k0 + wc8[i] * 8;
            whr[i] = *reinterpret_cast<const uint4*>(whi + off);
            wlr[i] = *reinterpret_cast<const uint4*>(wlo + off);
        }
    };
    auto store_b_lds = [&](int stage) {
#pragma unroll
        for (int i = 0; i < 2; ++i) {
            const int base = wrow[i] * LDS_STRIDE + wc8[i] * 8;
            *reinterpret_cast<uint4*>(&sBhi[stage][base]) = whr[i];
            *reinterpret_cast<uint4*>(&sBlo[stage][base]) = wlr[i];
        }
    };
#endif

    v8f acc[2][4];
#pragma unroll
    for (int mi = 0; mi < 2; ++mi)
#pragma unroll
        for (int ni = 0; ni < 4; ++ni)
            acc[mi][ni] = (v8f){0.f, 0.f, 0.f, 0.f, 0.f, 0.f, 0.f, 0.f};

    // ---- prologue ----
    int buf = 0;
    load_x_regs(0);
#if HAVE_ASYNC
    issue_async_b(0, 0);
#else
    load_b_regs(0);
#endif

    const int KSTEPS = INF / BK;  // 128
    for (int ks = 0; ks < KSTEPS; ++ks) {
        __syncthreads();          // all waves done computing step ks-1
        store_a_lds();            // A hi/lo -> LDS (single-buffered)

#if HAVE_ASYNC
        if (ks + 1 < KSTEPS) {
            issue_async_b((ks + 1) * BK, buf ^ 1);  // next slab -> other stage
            load_x_regs((ks + 1) * BK);             // overlap w/ WMMA below
            WAIT_ASYNC(4);  // 4 newest = next stage; current stage has landed
        } else {
            WAIT_ASYNC(0);
        }
#else
        store_b_lds(buf);
        if (ks + 1 < KSTEPS) {
            load_b_regs((ks + 1) * BK);
            load_x_regs((ks + 1) * BK);
        }
#endif
        __syncthreads();          // A stores + B slab visible to all waves

        // ---- compute: 24 WMMAs on tile (buf) ----
        v16bf ah[2], al[2];
#pragma unroll
        for (int mi = 0; mi < 2; ++mi) {
            const int arow = wm + mi * 16 + l16;
            ah[mi] = load_afrag(sAhi, arow, half);
            al[mi] = load_afrag(sAlo, arow, half);
        }
#pragma unroll
        for (int ni = 0; ni < 4; ++ni) {
            const int brow = wn + ni * 16 + l16;
            const v16bf bh = load_bfrag(sBhi[buf], brow, half);
            const v16bf bl = load_bfrag(sBlo[buf], brow, half);
#pragma unroll
            for (int mi = 0; mi < 2; ++mi) {
                acc[mi][ni] = __builtin_amdgcn_wmma_f32_16x16x32_bf16(
                    false, ah[mi], false, bh, (short)0, acc[mi][ni], false, false);
                acc[mi][ni] = __builtin_amdgcn_wmma_f32_16x16x32_bf16(
                    false, ah[mi], false, bl, (short)0, acc[mi][ni], false, false);
                acc[mi][ni] = __builtin_amdgcn_wmma_f32_16x16x32_bf16(
                    false, al[mi], false, bh, (short)0, acc[mi][ni], false, false);
            }
        }
        buf ^= 1;
    }

    // Epilogue: D element (m,n) lives at VGPR r with m = r + 8*half, n = l16.
#pragma unroll
    for (int ni = 0; ni < 4; ++ni) {
        const int col = nBase + wn + ni * 16 + l16;
        const float bb = bias[col];
#pragma unroll
        for (int mi = 0; mi < 2; ++mi) {
#pragma unroll
            for (int r = 0; r < 8; ++r) {
                const int row = mBase + wm + mi * 16 + r + 8 * half;
                out[(size_t)row * OUTF + col] = acc[mi][ni][r] + bb;
            }
        }
    }
}

// ---------------------------------------------------------------------------
extern "C" void kernel_launch(void* const* d_in, const int* in_sizes, int n_in,
                              void* d_out, int out_size, void* d_ws, size_t ws_size,
                              hipStream_t stream) {
    const float* x      = (const float*)d_in[0];   // [8192,4096] fp32
    const float* weight = (const float*)d_in[1];   // [65536] fp32 bins
    const float* bias   = (const float*)d_in[2];   // [4096] fp32
    const int*   idxh   = (const int*)d_in[3];     // [4096*4096] int32
    float* out = (float*)d_out;                    // [8192,4096] fp32

    // workspace: W_hi / W_lo bf16 planes, 32 MB each
    uint16_t* whi = (uint16_t*)d_ws;
    uint16_t* wlo = whi + (size_t)OUTF * INF;

    // 1) gather + split weights (16M elems, 4/thread)
    pack_weights<<<(OUTF * (size_t)INF) / (256 * 4), 256, 0, stream>>>(
        weight, idxh, whi, wlo);

    // 2) split-precision WMMA GEMM
    dim3 grid(OUTF / BN, TOKENS / BM);  // (32, 64)
    hashed_gemm<<<grid, 256, 0, stream>>>(x, whi, wlo, bias, out);
}
